// RNN_31104153158214
// MI455X (gfx1250) — compile-verified, matrix operand order checked by reading, not run on hardware
//
#include <hip/hip_runtime.h>
#include <hip/hip_bf16.h>

// Problem constants (match reference)
constexpr int kV = 5000;   // vocab
constexpr int kH = 512;    // hidden
constexpr int kN = 8;      // batch
constexpr int kT = 2048;   // seq len
constexpr int kVP = 5008;  // vocab padded to multiple of 16 for WMMA N tiles
constexpr int kHROW = 520; // LDS row stride (halves) to avoid bank conflicts

// Recurrence B-operand split: k-tiles 0..kJR-1 in VGPRs, kJR..15 in LDS.
constexpr int kJR = 7;                 // register-cached k-tiles per slice
constexpr int kJL = 16 - kJR;          // LDS-cached k-tiles per slice (9)

// Dynamic-LDS layout for the recurrence kernel (bytes)
constexpr int kTokOff  = 0;                      // int tokbuf[8]      (32 B)
constexpr int kHbufOff = 32;                     // f16 hbuf[16*kHROW] (16,640 B)
constexpr int kWhOff   = 32 + 16 * kHROW * 2;    // f16 whLds          (16,672)
constexpr int kNumFrag = 16 * 2 * kJL;           // 288 fragments
constexpr int kSmemBytes = kWhOff + kNumFrag * 512 * 2;  // 311,584 B (<320 KB)

typedef _Float16 h8_t __attribute__((ext_vector_type(8)));
typedef _Float16 v16h __attribute__((ext_vector_type(16)));
typedef float    v8f  __attribute__((ext_vector_type(8)));

// Load a 16-half WMMA fragment from two 16-byte chunks at p and p+pitch
// (halves), per the CDNA5 16-bit A/B layout: lanes 0-15 hold K=k..k+7 and
// k+16..k+23; lanes 16-31 hold K=k+8..k+15 and k+24..k+31.
__device__ __forceinline__ v16h load_frag(const _Float16* p, int pitchHalves) {
  h8_t lo = *(const h8_t*)(p);
  h8_t hi = *(const h8_t*)(p + pitchHalves);
  v16h r;
#pragma unroll
  for (int i = 0; i < 8; ++i) { r[i] = lo[i]; r[i + 8] = hi[i]; }
  return r;
}

__global__ void __launch_bounds__(256)
k_cvt_wh(const float* __restrict__ w, _Float16* __restrict__ o) {
  int i = blockIdx.x * 256 + threadIdx.x;      // kH*kH total
  o[i] = (_Float16)w[i];
}

__global__ void __launch_bounds__(256)
k_cvt_wo(const float* __restrict__ w, _Float16* __restrict__ o) {
  int i = blockIdx.x * 256 + threadIdx.x;      // kVP*kH total
  int row = i >> 9;                            // /kH
  o[i] = (row < kV) ? (_Float16)w[i] : (_Float16)0.f;
}

// --- Phase 1: sequential recurrence, one workgroup of 16 waves ----------
// Wh (f16, 512 KB) equals the whole WGP register file and exceeds the
// 320 KB LDS, so it is split: each wave owns two 16-column tiles; k-tiles
// 0..6 of its Wh^T slice live in VGPRs (14 frags = 112 regs), k-tiles
// 7..15 live in LDS pre-staged in fragment layout (288 KB).  The 2048-step
// steady state touches no global/scratch memory for the matmul.
__global__ void __launch_bounds__(512)
k_rnn_recur(const int* __restrict__ x, const float* __restrict__ E,
            const float* __restrict__ bh, const _Float16* __restrict__ wh16,
            _Float16* __restrict__ hall, float* __restrict__ hfinal) {
  extern __shared__ char smem[];
  int*      tokbuf = (int*)(smem + kTokOff);
  _Float16* hbuf   = (_Float16*)(smem + kHbufOff);
  _Float16* whLds  = (_Float16*)(smem + kWhOff);

  const int tid  = threadIdx.x;
  const int wave = tid >> 5;           // 0..15
  const int lane = tid & 31;
  const int lr   = lane & 15;
  const int hi   = lane >> 4;          // 0: D rows 0-7, 1: D rows 8-15
  const int ksoff = hi * 8;

  for (int i = tid; i < 16 * kHROW; i += 512) hbuf[i] = (_Float16)0.f;
  if (tid < 8) tokbuf[tid] = x[tid * kT];   // tokens for t=0

  const int n0 = wave * 32 + lr;       // tile-0 hidden column
  const int n1 = n0 + 16;              // tile-1 hidden column
  const float bh0 = bh[n0];
  const float bh1 = bh[n1];

  // Stage k-tiles kJR..15 of this wave's B slices into LDS, frag layout:
  // frag f = (wave*2+tile)*kJL + j at whLds[f*512 + c*256 + lane*8], c=0,1.
#pragma unroll
  for (int tile = 0; tile < 2; ++tile) {
    const int n = tile ? n1 : n0;
    const int f0 = (wave * 2 + tile) * kJL;
#pragma unroll
    for (int j = 0; j < kJL; ++j) {
      const _Float16* src = &wh16[(size_t)n * kH + (kJR + j) * 32 + ksoff];
      _Float16* dst = &whLds[(f0 + j) * 512 + lane * 8];
      *(h8_t*)(dst)       = *(const h8_t*)(src);
      *(h8_t*)(dst + 256) = *(const h8_t*)(src + 16);
    }
  }

  // Register-cache k-tiles 0..kJR-1 of both B slices (14 frags = 112 VGPRs).
  v16h bf0[kJR], bf1[kJR];
#pragma unroll
  for (int j = 0; j < kJR; ++j) {
    bf0[j] = load_frag(&wh16[(size_t)n0 * kH + j * 32 + ksoff], 16);
    bf1[j] = load_frag(&wh16[(size_t)n1 * kH + j * 32 + ksoff], 16);
  }
  __syncthreads();

  const int fb0 = (wave * 2 + 0) * kJL;   // LDS frag bases
  const int fb1 = (wave * 2 + 1) * kJL;

  for (int t = 0; t < kT; ++t) {
    // Opaque copies of the LDS frag bases: the asm makes them loop-variant
    // so the compiler cannot hoist the (loop-invariant) LDS B loads into
    // registers and spill them to scratch; loads stay ds_load_b128.
    int f0v = fb0, f1v = fb1;
    asm volatile("" : "+v"(f0v), "+v"(f1v));

    v8f c0 = {}; v8f c1 = {};
#pragma unroll
    for (int j = 0; j < kJR; ++j) {     // k-tiles 0..6: B from registers
      v16h a = load_frag(&hbuf[lr * kHROW + j * 32 + ksoff], 16);
      c0 = __builtin_amdgcn_wmma_f32_16x16x32_f16(false, a, false, bf0[j],
                                                  (short)0, c0, false, false);
      c1 = __builtin_amdgcn_wmma_f32_16x16x32_f16(false, a, false, bf1[j],
                                                  (short)0, c1, false, false);
    }
#pragma unroll
    for (int j = 0; j < kJL; ++j) {     // k-tiles 7..15: B from LDS
      v16h a  = load_frag(&hbuf[lr * kHROW + (kJR + j) * 32 + ksoff], 16);
      v16h b0 = load_frag(&whLds[(f0v + j) * 512 + lane * 8], 256);
      v16h b1 = load_frag(&whLds[(f1v + j) * 512 + lane * 8], 256);
      c0 = __builtin_amdgcn_wmma_f32_16x16x32_f16(false, a, false, b0,
                                                  (short)0, c0, false, false);
      c1 = __builtin_amdgcn_wmma_f32_16x16x32_f16(false, a, false, b1,
                                                  (short)0, c1, false, false);
    }
    int tk[8];
#pragma unroll
    for (int i = 0; i < 8; ++i) tk[i] = tokbuf[i];

    float r0[8], r1[8];
    if (hi == 0) {                       // lanes 0-15 hold batch rows 0-7
#pragma unroll
      for (int i = 0; i < 8; ++i) {
        r0[i] = tanhf(c0[i] + bh0 + E[(size_t)tk[i] * kH + n0]);
        r1[i] = tanhf(c1[i] + bh1 + E[(size_t)tk[i] * kH + n1]);
        hall[((size_t)t * kN + i) * kH + n0] = (_Float16)r0[i];
        hall[((size_t)t * kN + i) * kH + n1] = (_Float16)r1[i];
      }
      if (t == kT - 1) {
#pragma unroll
        for (int i = 0; i < 8; ++i) {
          hfinal[i * kH + n0] = r0[i];
          hfinal[i * kH + n1] = r1[i];
        }
      }
    }
    __syncthreads();                     // all reads of h_{t-1} done
    if (hi == 0) {
#pragma unroll
      for (int i = 0; i < 8; ++i) {
        hbuf[i * kHROW + n0] = (_Float16)r0[i];
        hbuf[i * kHROW + n1] = (_Float16)r1[i];
      }
    }
    if (tid < 8 && t + 1 < kT) tokbuf[tid] = x[tid * kT + t + 1];
    __syncthreads();                     // h_t visible to everyone
  }
}

// --- Phase 2: Z = H_all @ Wo^T + bo, written transposed to [N,T,V] ------
__global__ void __launch_bounds__(256)
k_proj(const _Float16* __restrict__ hall, const _Float16* __restrict__ wo16,
       const float* __restrict__ bo, float* __restrict__ out) {
  const int tid  = threadIdx.x;
  const int wave = tid >> 5;
  const int lane = tid & 31;
  const int lr   = lane & 15;
  const int hi   = lane >> 4;
  const int ksoff = hi * 8;
  const int r0   = blockIdx.x * 16;                // row block of H_all
  const int col0 = (blockIdx.y * 8 + wave) * 16;   // N tile (wave-uniform)
  if (col0 >= kVP) return;

  v16h afrag[16];
#pragma unroll
  for (int kt = 0; kt < 16; ++kt)
    afrag[kt] = load_frag(&hall[(size_t)(r0 + lr) * kH + kt * 32 + ksoff], 16);

  const int n = col0 + lr;
  v8f acc = {};
#pragma unroll
  for (int kt = 0; kt < 16; ++kt) {
    v16h b = load_frag(&wo16[(size_t)n * kH + kt * 32 + ksoff], 16);
    acc = __builtin_amdgcn_wmma_f32_16x16x32_f16(false, afrag[kt], false, b,
                                                 (short)0, acc, false, false);
  }
  if (n < kV) {
    const float bn = bo[n];
#pragma unroll
    for (int i = 0; i < 8; ++i) {
      const int r  = r0 + i + 8 * hi;    // global row = t*N + batch
      const int tt = r >> 3;
      const int bb = r & 7;
      out[((size_t)bb * kT + tt) * kV + n] = acc[i] + bn;
    }
  }
}

// --- Phase 3: row softmax in place (one block per [n,t] row) ------------
__global__ void __launch_bounds__(256)
k_softmax(float* __restrict__ out) {
  __shared__ float red[256];
  const int tid = threadIdx.x;
  float* p = out + (size_t)blockIdx.x * kV;
  float vals[20];
  int cnt = 0;
  float m = -3.4e38f;
  for (int v = tid; v < kV; v += 256) { float z = p[v]; vals[cnt++] = z; m = fmaxf(m, z); }
  red[tid] = m; __syncthreads();
  for (int s = 128; s > 0; s >>= 1) {
    if (tid < s) red[tid] = fmaxf(red[tid], red[tid + s]);
    __syncthreads();
  }
  m = red[0]; __syncthreads();
  float sum = 0.f;
  for (int j = 0; j < cnt; ++j) { vals[j] = __expf(vals[j] - m); sum += vals[j]; }
  red[tid] = sum; __syncthreads();
  for (int s = 128; s > 0; s >>= 1) {
    if (tid < s) red[tid] += red[tid + s];
    __syncthreads();
  }
  const float inv = 1.f / red[0];
  int j = 0;
  for (int v = tid; v < kV; v += 256) p[v] = vals[j++] * inv;
}

extern "C" void kernel_launch(void* const* d_in, const int* in_sizes, int n_in,
                              void* d_out, int out_size, void* d_ws, size_t ws_size,
                              hipStream_t stream) {
  const int*   x  = (const int*)d_in[0];
  const float* E  = (const float*)d_in[1];
  const float* Wh = (const float*)d_in[2];
  const float* bh = (const float*)d_in[3];
  const float* Wo = (const float*)d_in[4];
  const float* bo = (const float*)d_in[5];
  float* out = (float*)d_out;

  // workspace layout (f16): Wh (512KB) | Wo padded (5.0MB) | H_all (16MB)
  _Float16* wh16 = (_Float16*)d_ws;
  _Float16* wo16 = wh16 + (size_t)kH * kH;
  _Float16* hall = wo16 + (size_t)kVP * kH;

  k_cvt_wh<<<(kH * kH) / 256, 256, 0, stream>>>(Wh, wh16);
  k_cvt_wo<<<(kVP * kH) / 256, 256, 0, stream>>>(Wo, wo16);

  k_rnn_recur<<<1, 512, kSmemBytes, stream>>>(x, E, bh, wh16, hall,
                                              out + (size_t)kN * kT * kV);

  dim3 gproj(16384 / 16, (kVP / 16 + 7) / 8);   // 1024 x 40
  k_proj<<<gproj, 256, 0, stream>>>(hall, wo16, bo, out);

  k_softmax<<<kN * kT, 256, 0, stream>>>(out);
}
